// DCNv3_6502580486619
// MI455X (gfx1250) — matrix operand order repaired and checked by reading, not compiled, not run on hardware
//
#include <hip/hip_runtime.h>
#include <hip/hip_bf16.h>

// ---------------------------------------------------------------------------
// DCNv3 forward, MI455X (gfx1250, wave32, WMMA 16x16x32 bf16)
// N=4, H=W=64, C=256, G=8, GC=32, K=3, P=9, PAD=1, Hi=Wi=66
// ---------------------------------------------------------------------------

#define Nn   4
#define Hh   64
#define Ww   64
#define Cc   256
#define Gg   8
#define GC   32
#define Pp   9
#define HiP  66          // H + 2*PAD
#define WiP  66
#define Mtot (Nn*Hh*Ww)  // 16384 rows for all GEMMs

typedef __attribute__((ext_vector_type(16))) __bf16 v16bf;
typedef __attribute__((ext_vector_type(8)))  __bf16 v8bf;
typedef __attribute__((ext_vector_type(8)))  float  v8f;

__device__ __forceinline__ __bf16 f2bf(float f) {
  union { float f; unsigned u; } v; v.f = f;
  unsigned r = v.u + 0x7FFFu + ((v.u >> 16) & 1u);   // round-to-nearest-even
  unsigned short h = (unsigned short)(r >> 16);
  __bf16 out; __builtin_memcpy(&out, &h, 2);
  return out;
}

// --------------------------- elementwise helpers ---------------------------

__global__ void k_f32_to_bf16(const float* __restrict__ in, __bf16* __restrict__ out, int n) {
  int i = blockIdx.x * blockDim.x + threadIdx.x;
  if (i < n) out[i] = f2bf(in[i]);
}

__global__ void k_fill_zero_f32(float* __restrict__ p, int n) {
  int i = blockIdx.x * blockDim.x + threadIdx.x;
  if (i < n) p[i] = 0.0f;
}

// W is (K x N) row-major fp32; out is (Npad x K) row-major bf16 (= B^T, zero-padded rows)
__global__ void k_transpose_to_bf16(const float* __restrict__ W, __bf16* __restrict__ out,
                                    int K, int N, int Npad) {
  int i = blockIdx.x * blockDim.x + threadIdx.x;
  if (i >= Npad * K) return;
  int nrow = i / K;
  int k    = i - nrow * K;
  float v = (nrow < N) ? W[(size_t)k * N + nrow] : 0.0f;
  out[i] = f2bf(v);
}

// depthwise 3x3 conv (pad 1) + BatchNorm(eval) + SiLU, NHWC in, bf16 out
__global__ void k_dwconv_bn_silu(const float* __restrict__ x, const float* __restrict__ dww,
                                 const float* __restrict__ gamma, const float* __restrict__ beta,
                                 const float* __restrict__ mean,  const float* __restrict__ var,
                                 __bf16* __restrict__ x1) {
  int idx = blockIdx.x * blockDim.x + threadIdx.x;
  if (idx >= Mtot * Cc) return;
  int c  = idx & (Cc - 1);
  int w  = (idx >> 8) & 63;
  int h  = (idx >> 14) & 63;
  int n  = idx >> 20;
  float acc = 0.0f;
#pragma unroll
  for (int kh = 0; kh < 3; ++kh) {
    int hy = h + kh - 1;
    if ((unsigned)hy >= (unsigned)Hh) continue;
#pragma unroll
    for (int kw = 0; kw < 3; ++kw) {
      int wx = w + kw - 1;
      if ((unsigned)wx >= (unsigned)Ww) continue;
      acc += x[(((size_t)n * Hh + hy) * Ww + wx) * Cc + c] * dww[c * 9 + kh * 3 + kw];
    }
  }
  float sc = gamma[c] * __frsqrt_rn(var[c] + 1e-3f);
  float y  = (acc - mean[c]) * sc + beta[c];
  float s  = y / (1.0f + __expf(-y));     // SiLU
  x1[idx] = f2bf(s);
}

// softmax over the 9 taps of each (pos, group); in-place on (Mtot x 72) fp32
__global__ void k_softmax_mask(float* __restrict__ mask) {
  int t = blockIdx.x * blockDim.x + threadIdx.x;
  if (t >= Mtot * Gg) return;
  float* p = mask + (size_t)(t >> 3) * (Gg * Pp) + (t & 7) * Pp;
  float v[Pp];
  float m = -1e30f;
#pragma unroll
  for (int i = 0; i < Pp; ++i) { v[i] = p[i]; m = fmaxf(m, v[i]); }
  float s = 0.0f;
#pragma unroll
  for (int i = 0; i < Pp; ++i) { v[i] = __expf(v[i] - m); s += v[i]; }
  float inv = 1.0f / s;
#pragma unroll
  for (int i = 0; i < Pp; ++i) p[i] = v[i] * inv;
}

// --------------------------- WMMA GEMM ---------------------------
// D(M x Nreal) = A(M x K, bf16) * B(K x Nreal, via Bt = B^T padded to Npad rows) + bias
// One wave computes a 16(M) x 64(N) strip: A fragment reused across 4 WMMA tiles.
// store_mode 0: plain row-major fp32 (ld = Nreal)
// store_mode 1: scatter into padded NHWC fp32 buffer (Nn,HiP,WiP,Cc), row m -> (n,h+1,w+1)
__global__ void k_gemm_wmma_bf16(const __bf16* __restrict__ A, const __bf16* __restrict__ Bt,
                                 const float* __restrict__ bias, float* __restrict__ out,
                                 int M, int Nreal, int Npad, int K, int store_mode) {
  const int lane = threadIdx.x & 31;
  const int wid  = (blockIdx.x * blockDim.x + threadIdx.x) >> 5;
  const int nquads = Npad >> 6;
  const int mtiles = M >> 4;
  if (wid >= mtiles * nquads) return;           // wave-uniform: EXEC stays all-1 for WMMA
  const int mtile = wid / nquads;
  const int nq    = wid - mtile * nquads;

  const int l15   = lane & 15;
  const int khalf = lane >> 4;                  // 0: K-lo half, 1: K-hi half (contiguous k order chosen here)
  const int row   = (mtile << 4) + l15;
  const int acol  = khalf * 16;
  const int bklo  = khalf * 16;
  const int bcol0 = (nq << 6) + l15;

  const __bf16* __restrict__ Arow = A + (size_t)row * K;

  v8f acc[4] = {v8f{}, v8f{}, v8f{}, v8f{}};

#pragma unroll
  for (int kb = 0; kb < 256; kb += 32) {        // K == 256 for every GEMM here
    // A fragment: lane holds row `row`, K = kb+acol..+15 (contiguous, matches B)
    v8bf lo = *(const v8bf*)(Arow + kb + acol);
    v8bf hi = *(const v8bf*)(Arow + kb + acol + 8);
    v16bf a;
#pragma unroll
    for (int i = 0; i < 8; ++i) { a[i] = lo[i]; a[8 + i] = hi[i]; }
    // B fragments: lane holds 16 consecutive K of column bcol0 + t*16 (from B^T rows)
#pragma unroll
    for (int t = 0; t < 4; ++t) {
      v16bf b = *(const v16bf*)(Bt + (size_t)(bcol0 + t * 16) * K + kb + bklo);
      acc[t] = __builtin_amdgcn_wmma_f32_16x16x32_bf16(
          /*neg_a=*/false, a, /*neg_b=*/false, b,
          /*c_mod=*/(short)0, acc[t], /*reuse_a=*/false, /*reuse_b=*/false);
    }
  }

  // C/D layout: lanes 0-15 -> M = j, lanes 16-31 -> M = 8+j; N = lane&15 (per tile)
#pragma unroll
  for (int t = 0; t < 4; ++t) {
    int n = (nq << 6) + t * 16 + l15;
    if (n < Nreal) {
      float bv = bias ? bias[n] : 0.0f;
#pragma unroll
      for (int j = 0; j < 8; ++j) {
        int m = (mtile << 4) + khalf * 8 + j;
        float v = acc[t][j] + bv;
        if (store_mode == 0) {
          out[(size_t)m * Nreal + n] = v;
        } else {
          int w = m & 63, h = (m >> 6) & 63, ni = m >> 12;
          out[(((size_t)ni * HiP + (h + 1)) * WiP + (w + 1)) * Cc + n] = v;
        }
      }
    }
  }
}

// --------------------------- DCNv3 sampling core ---------------------------
// One wave per (pos, group); lane = channel within group (GC == 32 == wave32).
// Sample coords in padded 66x66 space: sx = w + p/3 + off_x, sy = h + p%3 + off_y
// (exactly the reference's ref + kgrid + off after denormalization).
__global__ void k_dcnv3_core(const float* __restrict__ xpad,   // (Nn,66,66,Cc)
                             const float* __restrict__ offs,   // (Mtot,144)
                             const float* __restrict__ mask,   // (Mtot,72) softmaxed
                             __bf16* __restrict__ core) {      // (Mtot,Cc) bf16
  const int lane = threadIdx.x & 31;
  const int wid  = (blockIdx.x * blockDim.x + threadIdx.x) >> 5;
  if (wid >= Mtot * Gg) return;
  const int g   = wid & 7;
  const int pos = wid >> 3;
  const int w   = pos & 63;
  const int h   = (pos >> 6) & 63;
  const int n   = pos >> 12;

  const float* __restrict__ off = offs + (size_t)pos * (Gg * Pp * 2) + g * (Pp * 2);
  const float* __restrict__ mk  = mask + (size_t)pos * (Gg * Pp)     + g * Pp;
  const float* __restrict__ img = xpad + (size_t)n * (HiP * WiP * Cc) + g * GC + lane;

  float acc = 0.0f;
#pragma unroll
  for (int p = 0; p < Pp; ++p) {
    float sx = (float)(w + p / 3) + off[2 * p + 0];
    float sy = (float)(h + p % 3) + off[2 * p + 1];
    float fx = floorf(sx), fy = floorf(sy);
    int   x0 = (int)fx,    y0 = (int)fy;
    float tx = sx - fx,    ty = sy - fy;
    float wgt = mk[p];
    float cw[4] = { (1.0f - tx) * (1.0f - ty) * wgt, tx * (1.0f - ty) * wgt,
                    (1.0f - tx) * ty * wgt,          tx * ty * wgt };
    int   cx[4] = { x0, x0 + 1, x0,     x0 + 1 };
    int   cy[4] = { y0, y0,     y0 + 1, y0 + 1 };
#pragma unroll
    for (int q = 0; q < 4; ++q) {
      if ((unsigned)cx[q] < (unsigned)WiP && (unsigned)cy[q] < (unsigned)HiP)
        acc += cw[q] * img[(((size_t)cy[q] * WiP) + cx[q]) << 8];   // *Cc
    }
  }
  core[(size_t)pos * Cc + g * GC + lane] = f2bf(acc);
}

// --------------------------- launcher ---------------------------

extern "C" void kernel_launch(void* const* d_in, const int* in_sizes, int n_in,
                              void* d_out, int out_size, void* d_ws, size_t ws_size,
                              hipStream_t stream) {
  const float* x        = (const float*)d_in[0];
  const float* dw_w     = (const float*)d_in[1];
  const float* bn_gamma = (const float*)d_in[2];
  const float* bn_beta  = (const float*)d_in[3];
  const float* bn_mean  = (const float*)d_in[4];
  const float* bn_var   = (const float*)d_in[5];
  const float* W_off    = (const float*)d_in[6];
  const float* b_off    = (const float*)d_in[7];
  const float* W_mask   = (const float*)d_in[8];
  const float* b_mask   = (const float*)d_in[9];
  const float* W_in     = (const float*)d_in[10];
  const float* b_in     = (const float*)d_in[11];
  const float* W_out    = (const float*)d_in[12];
  const float* b_out    = (const float*)d_in[13];
  float* out = (float*)d_out;

  char* ws = (char*)d_ws;
  size_t cur = 0;
  auto alloc = [&](size_t bytes) -> void* {
    void* p = ws + cur;
    cur = (cur + bytes + 255) & ~(size_t)255;
    return p;
  };

  const size_t NE = (size_t)Mtot * Cc;                 // 4,194,304 elems
  __bf16* xbf    = (__bf16*)alloc(NE * 2);
  __bf16* x1bf   = (__bf16*)alloc(NE * 2);
  float*  xpad   = (float*) alloc((size_t)Nn * HiP * WiP * Cc * 4);
  float*  offset = (float*) alloc((size_t)Mtot * (Gg * Pp * 2) * 4);
  float*  maskb  = (float*) alloc((size_t)Mtot * (Gg * Pp) * 4);
  __bf16* corebf = (__bf16*)alloc(NE * 2);
  __bf16* Win_t  = (__bf16*)alloc((size_t)256 * 256 * 2);
  __bf16* Woff_t = (__bf16*)alloc((size_t)192 * 256 * 2);
  __bf16* Wmsk_t = (__bf16*)alloc((size_t)128 * 256 * 2);
  __bf16* Wout_t = (__bf16*)alloc((size_t)256 * 256 * 2);
  (void)ws_size; (void)n_in; (void)in_sizes; (void)out_size;

  const int TB = 256;                                  // 8 wave32 per block
  // 1) convert inputs / weights to bf16 (weights transposed + zero-padded)
  k_f32_to_bf16<<<(int)((NE + TB - 1) / TB), TB, 0, stream>>>(x, xbf, (int)NE);
  k_transpose_to_bf16<<<(256 * 256 + TB - 1) / TB, TB, 0, stream>>>(W_in,   Win_t,  256, 256, 256);
  k_transpose_to_bf16<<<(192 * 256 + TB - 1) / TB, TB, 0, stream>>>(W_off,  Woff_t, 256, 144, 192);
  k_transpose_to_bf16<<<(128 * 256 + TB - 1) / TB, TB, 0, stream>>>(W_mask, Wmsk_t, 256,  72, 128);
  k_transpose_to_bf16<<<(256 * 256 + TB - 1) / TB, TB, 0, stream>>>(W_out,  Wout_t, 256, 256, 256);

  // 2) fused depthwise conv + BN + SiLU -> x1 (bf16)
  k_dwconv_bn_silu<<<(int)((NE + TB - 1) / TB), TB, 0, stream>>>(
      x, dw_w, bn_gamma, bn_beta, bn_mean, bn_var, x1bf);

  // 3) x_proj GEMM -> padded sampling buffer (border zeroed first)
  const int npadE = Nn * HiP * WiP * Cc;
  k_fill_zero_f32<<<(npadE + TB - 1) / TB, TB, 0, stream>>>(xpad, npadE);
  {
    int waves = (Mtot / 16) * (256 / 64);              // 4096
    k_gemm_wmma_bf16<<<waves / 8, TB, 0, stream>>>(xbf, Win_t, b_in, xpad,
                                                   Mtot, 256, 256, 256, /*padded*/1);
  }
  // 4) offset / mask GEMMs from x1
  {
    int waves = (Mtot / 16) * (192 / 64);              // 3072
    k_gemm_wmma_bf16<<<waves / 8, TB, 0, stream>>>(x1bf, Woff_t, b_off, offset,
                                                   Mtot, 144, 192, 256, 0);
  }
  {
    int waves = (Mtot / 16) * (128 / 64);              // 2048
    k_gemm_wmma_bf16<<<waves / 8, TB, 0, stream>>>(x1bf, Wmsk_t, b_mask, maskb,
                                                   Mtot, 72, 128, 256, 0);
  }
  // 5) per-(pos,group) softmax over 9 taps
  k_softmax_mask<<<(Mtot * Gg + TB - 1) / TB, TB, 0, stream>>>(maskb);

  // 6) deformable sampling core (1 wave per (pos,group), lane = channel)
  k_dcnv3_core<<<(Mtot * Gg) / 8, TB, 0, stream>>>(xpad, offset, maskb, corebf);

  // 7) output projection GEMM -> d_out (fp32)
  {
    int waves = (Mtot / 16) * (256 / 64);              // 4096
    k_gemm_wmma_bf16<<<waves / 8, TB, 0, stream>>>(corebf, Wout_t, b_out, out,
                                                   Mtot, 256, 256, 256, 0);
  }
}
